// PointerGenerator_39857296507386
// MI455X (gfx1250) — matrix operand order repaired and checked by reading, not compiled, not run on hardware
//
#include <hip/hip_runtime.h>
#include <hip/hip_bf16.h>

#define BB 8
#define TT 128
#define TS 512
#define DD 512
#define VV 32000

typedef __attribute__((ext_vector_type(16))) __bf16        v16bf;
typedef __attribute__((ext_vector_type(8)))  float         v8f;
typedef __attribute__((ext_vector_type(4)))  unsigned int  u32x4;

union Frag { v16bf v; u32x4 q[2]; };

__device__ __forceinline__ unsigned short f2bf(float f) {
  unsigned int u = __float_as_uint(f);
  u += 0x7FFFu + ((u >> 16) & 1u);          // round-to-nearest-even
  return (unsigned short)(u >> 16);
}

// ---------------------------------------------------------------------------
// Generic batched GEMM: C[b][M,N] = A[b][M,K](bf16) * B[b][N,K]^T(bf16) + bias
// One wave computes a 16x64 tile: 1 A-fragment x 4 B-fragments per K-step of 32.
// Fragments use contiguous k order (k = 16*half + s) in both A and B.
// ---------------------------------------------------------------------------
__global__ __launch_bounds__(128) void gemm_bf16_tn(
    const unsigned short* __restrict__ A,
    const unsigned short* __restrict__ B,
    const float* __restrict__ bias,
    float* __restrict__ C,
    int M, int N, int K,
    long long sA, long long sB, long long sC)
{
  const int lane = threadIdx.x & 31;
  const int wid  = blockIdx.x * (blockDim.x >> 5) + (threadIdx.x >> 5);
  const int tilesN = N >> 6;
  const int tilesM = M >> 4;
  if (wid >= tilesM * tilesN) return;          // wave-uniform guard
  const int wm = wid / tilesN;
  const int wn = wid % tilesN;

  const unsigned short* Ab = A + (long long)blockIdx.y * sA;
  const unsigned short* Bb = B + (long long)blockIdx.y * sB;
  float* Cb = C + (long long)blockIdx.y * sC;

  const int hi = lane >> 4;                    // half-wave select
  const int ln = lane & 15;
  // A: lane row = wm*16+ln; lanes 16..31 shift K by +16
  const unsigned short* aRow  = Ab + (long long)((wm << 4) + ln) * K + (hi << 4);
  // B: lane col = wn*64+ln(+16*nsub); lanes 16..31 shift K by +16
  const unsigned short* bRow0 = Bb + (long long)((wn << 6) + ln) * K + (hi << 4);
  const long long bStride16 = 16ll * K;

  v8f c0 = {}, c1 = {}, c2 = {}, c3 = {};

  for (int kk = 0; kk < K; kk += 32) {
    Frag a;
    a.q[0] = *(const u32x4*)(aRow + kk);        // K=kk..kk+7   (+16 hi)
    a.q[1] = *(const u32x4*)(aRow + kk + 8);    // K=kk+8..15   (+16 hi)

    Frag b0, b1, b2, b3;
    const unsigned short* bp = bRow0 + kk;
    b0.q[0] = *(const u32x4*)(bp);  b0.q[1] = *(const u32x4*)(bp + 8);
    bp += bStride16;
    b1.q[0] = *(const u32x4*)(bp);  b1.q[1] = *(const u32x4*)(bp + 8);
    bp += bStride16;
    b2.q[0] = *(const u32x4*)(bp);  b2.q[1] = *(const u32x4*)(bp + 8);
    bp += bStride16;
    b3.q[0] = *(const u32x4*)(bp);  b3.q[1] = *(const u32x4*)(bp + 8);

    c0 = __builtin_amdgcn_wmma_f32_16x16x32_bf16(false, a.v, false, b0.v, (short)0, c0, false, false);
    c1 = __builtin_amdgcn_wmma_f32_16x16x32_bf16(false, a.v, false, b1.v, (short)0, c1, false, false);
    c2 = __builtin_amdgcn_wmma_f32_16x16x32_bf16(false, a.v, false, b2.v, (short)0, c2, false, false);
    c3 = __builtin_amdgcn_wmma_f32_16x16x32_bf16(false, a.v, false, b3.v, (short)0, c3, false, false);
  }

  // C/D layout: VGPR r -> M = r (+8 for lanes 16..31); N = lane&15
  const int mBase = (wm << 4) + (hi << 3);
  const int n0 = (wn << 6) + ln;
  const float bv0 = bias ? bias[n0]      : 0.f;
  const float bv1 = bias ? bias[n0 + 16] : 0.f;
  const float bv2 = bias ? bias[n0 + 32] : 0.f;
  const float bv3 = bias ? bias[n0 + 48] : 0.f;
#pragma unroll
  for (int r = 0; r < 8; ++r) {
    float* crow = Cb + (long long)(mBase + r) * N;
    crow[n0]      = c0[r] + bv0;
    crow[n0 + 16] = c1[r] + bv1;
    crow[n0 + 32] = c2[r] + bv2;
    crow[n0 + 48] = c3[r] + bv3;
  }
}

// ---------------------------------------------------------------------------
__global__ void conv_f32_bf16(const float* __restrict__ in,
                              unsigned short* __restrict__ out, int n) {
  int i = blockIdx.x * blockDim.x + threadIdx.x;
  if (i < n) out[i] = f2bf(in[i]);
}

// encoded_query [B,TS,DD] -> eqT bf16 [B,DD,TS]
__global__ void conv_eqT_bf16(const float* __restrict__ in,
                              unsigned short* __restrict__ out) {
  int i = blockIdx.x * blockDim.x + threadIdx.x;   // B*TS*DD
  int b   = i >> 18;
  int rem = i & 262143;
  int d   = rem >> 9;
  int s   = rem & 511;
  out[i] = f2bf(in[(b << 18) + (s << 9) + d]);
}

__global__ void zero_kernel(float* __restrict__ p, int n4) {
  int i = blockIdx.x * blockDim.x + threadIdx.x;
  if (i < n4) ((float4*)p)[i] = make_float4(0.f, 0.f, 0.f, 0.f);
}

// masked scale+softmax over T_SRC=512; writes normalized attn (f32) in place
// and a bf16 copy for the text_vec GEMM.
__global__ __launch_bounds__(256) void attn_softmax_kernel(
    float* __restrict__ scores, unsigned short* __restrict__ attn_bf,
    const int* __restrict__ qmask)
{
  const int bt = blockIdx.x;                 // b*TT + t
  const int b  = bt >> 7;
  float* row = scores + (long long)bt * TS;
  const int* mrow = qmask + b * TS;
  const float scale = 0.04419417382415922f;  // 1/sqrt(512)
  const int t0 = threadIdx.x;

  float x0 = row[t0]       * scale; if (mrow[t0]       <= 0) x0 = -1e9f;
  float x1 = row[t0 + 256] * scale; if (mrow[t0 + 256] <= 0) x1 = -1e9f;

  __shared__ float red[256];
  red[t0] = fmaxf(x0, x1); __syncthreads();
  for (int off = 128; off > 0; off >>= 1) {
    if (t0 < off) red[t0] = fmaxf(red[t0], red[t0 + off]);
    __syncthreads();
  }
  const float m = red[0]; __syncthreads();

  float e0 = __expf(x0 - m), e1 = __expf(x1 - m);
  red[t0] = e0 + e1; __syncthreads();
  for (int off = 128; off > 0; off >>= 1) {
    if (t0 < off) red[t0] += red[t0 + off];
    __syncthreads();
  }
  const float inv = 1.0f / red[0];
  const float a0 = e0 * inv, a1 = e1 * inv;
  row[t0] = a0; row[t0 + 256] = a1;
  attn_bf[(long long)bt * TS + t0]       = f2bf(a0);
  attn_bf[(long long)bt * TS + t0 + 256] = f2bf(a1);
}

// single-pass online softmax stats (max, sumexp) per vocab row
__global__ __launch_bounds__(256) void vocab_stats_kernel(
    const float* __restrict__ logits, float* __restrict__ rmax,
    float* __restrict__ rsum)
{
  const int row = blockIdx.x;
  const float* p = logits + (long long)row * VV;
  float m = -3.0e38f, s = 0.f;
  for (int i = threadIdx.x; i < VV; i += 256) {
    float x = p[i];
    if (x > m) { s *= __expf(m - x); m = x; }
    s += __expf(x - m);
  }
  __shared__ float sm[256], ss[256];
  sm[threadIdx.x] = m; ss[threadIdx.x] = s; __syncthreads();
  for (int off = 128; off > 0; off >>= 1) {
    if (threadIdx.x < off) {
      float m1 = sm[threadIdx.x],       s1 = ss[threadIdx.x];
      float m2 = sm[threadIdx.x + off], s2 = ss[threadIdx.x + off];
      float mn = fmaxf(m1, m2);
      sm[threadIdx.x] = mn;
      ss[threadIdx.x] = s1 * __expf(m1 - mn) + s2 * __expf(m2 - mn);
    }
    __syncthreads();
  }
  if (threadIdx.x == 0) { rmax[row] = sm[0]; rsum[row] = ss[0]; }
}

__global__ __launch_bounds__(256) void switch_kernel(
    const float* __restrict__ x, const float* __restrict__ tv,
    const float* __restrict__ etgt, const float* __restrict__ Wg,
    const float* __restrict__ bg, float* __restrict__ sw)
{
  const int bt = blockIdx.x;
  const float* xr = x    + (long long)bt * DD;
  const float* tr = tv   + (long long)bt * DD;
  const float* er = etgt + (long long)bt * DD;
  float acc = 0.f;
  for (int i = threadIdx.x; i < DD; i += 256)
    acc += xr[i] * Wg[i] + tr[i] * Wg[DD + i] + er[i] * Wg[2 * DD + i];
  __shared__ float red[256];
  red[threadIdx.x] = acc; __syncthreads();
  for (int off = 128; off > 0; off >>= 1) {
    if (threadIdx.x < off) red[threadIdx.x] += red[threadIdx.x + off];
    __syncthreads();
  }
  if (threadIdx.x == 0) sw[bt] = 1.0f / (1.0f + __expf(-(red[0] + bg[0])));
}

__global__ void scatter_kernel(const float* __restrict__ attn,
                               const int* __restrict__ query,
                               float* __restrict__ ptrP)
{
  int i = blockIdx.x * blockDim.x + threadIdx.x;   // B*TT*TS
  int s  = i & (TS - 1);
  int bt = i >> 9;
  int b  = bt >> 7;
  int v  = query[b * TS + s];
  atomicAdd(ptrP + (long long)bt * VV + v, attn[i]);
}

// fused: p_vocab = exp(l - max)/sum ; out = log(sw*p_vocab + (1-sw)*p_ptr + 1e-30)
__global__ __launch_bounds__(256) void combine_kernel(
    float* __restrict__ out, const float* __restrict__ ptrP,
    const float* __restrict__ rmax, const float* __restrict__ rsum,
    const float* __restrict__ sw)
{
  long long i = ((long long)blockIdx.x * blockDim.x + threadIdx.x) * 4;
  int bt = (int)(i / VV);
  const float mx  = rmax[bt];
  const float inv = 1.0f / rsum[bt];
  const float s   = sw[bt];
  const float t   = 1.0f - s;
  float4 l  = *(const float4*)(out + i);
  float4 pp = *(const float4*)(ptrP + i);
  float4 r;
  r.x = __logf(s * __expf(l.x - mx) * inv + t * pp.x + 1e-30f);
  r.y = __logf(s * __expf(l.y - mx) * inv + t * pp.y + 1e-30f);
  r.z = __logf(s * __expf(l.z - mx) * inv + t * pp.z + 1e-30f);
  r.w = __logf(s * __expf(l.w - mx) * inv + t * pp.w + 1e-30f);
  *(float4*)(out + i) = r;
}

// ---------------------------------------------------------------------------
extern "C" void kernel_launch(void* const* d_in, const int* in_sizes, int n_in,
                              void* d_out, int out_size, void* d_ws, size_t ws_size,
                              hipStream_t stream) {
  (void)in_sizes; (void)n_in; (void)out_size; (void)ws_size;

  const float* x    = (const float*)d_in[0];
  const int*   query= (const int*)  d_in[1];
  const float* eq   = (const float*)d_in[2];
  const int*   qmask= (const int*)  d_in[3];
  const float* etgt = (const float*)d_in[4];
  const float* vg   = (const float*)d_in[5];
  const float* Wq   = (const float*)d_in[6];
  const float* bq   = (const float*)d_in[7];
  const float* Wk   = (const float*)d_in[8];
  const float* bk   = (const float*)d_in[9];
  const float* Wg   = (const float*)d_in[10];
  const float* bg   = (const float*)d_in[11];
  float* out = (float*)d_out;

  char* ws = (char*)d_ws;
  size_t o = 0;
  auto alloc = [&](size_t bytes) { size_t r = o; o += (bytes + 255) & ~size_t(255); return r; };
  const size_t oPTR   = alloc(4ull * BB * TT * VV);
  const size_t oVGBF  = alloc(2ull * VV * DD);
  const size_t oXBF   = alloc(2ull * BB * TT * DD);
  const size_t oEQBF  = alloc(2ull * BB * TS * DD);
  const size_t oEQTBF = alloc(2ull * BB * TS * DD);
  const size_t oWQBF  = alloc(2ull * DD * DD);
  const size_t oWKBF  = alloc(2ull * DD * DD);
  const size_t oQF    = alloc(4ull * BB * TT * DD);
  const size_t oKF    = alloc(4ull * BB * TS * DD);
  const size_t oQBF   = alloc(2ull * BB * TT * DD);
  const size_t oKBF   = alloc(2ull * BB * TS * DD);
  const size_t oSC    = alloc(4ull * BB * TT * TS);
  const size_t oABF   = alloc(2ull * BB * TT * TS);
  const size_t oTV    = alloc(4ull * BB * TT * DD);
  const size_t oSW    = alloc(4ull * BB * TT);
  const size_t oRMAX  = alloc(4ull * BB * TT);
  const size_t oRSUM  = alloc(4ull * BB * TT);

  float*          ptrP  = (float*)(ws + oPTR);
  unsigned short* vgbf  = (unsigned short*)(ws + oVGBF);
  unsigned short* xbf   = (unsigned short*)(ws + oXBF);
  unsigned short* eqbf  = (unsigned short*)(ws + oEQBF);
  unsigned short* eqtbf = (unsigned short*)(ws + oEQTBF);
  unsigned short* wqbf  = (unsigned short*)(ws + oWQBF);
  unsigned short* wkbf  = (unsigned short*)(ws + oWKBF);
  float*          qf    = (float*)(ws + oQF);
  float*          kf    = (float*)(ws + oKF);
  unsigned short* qbf   = (unsigned short*)(ws + oQBF);
  unsigned short* kbf   = (unsigned short*)(ws + oKBF);
  float*          sc    = (float*)(ws + oSC);
  unsigned short* abf   = (unsigned short*)(ws + oABF);
  float*          tv    = (float*)(ws + oTV);
  float*          sw    = (float*)(ws + oSW);
  float*          rmax  = (float*)(ws + oRMAX);
  float*          rsum  = (float*)(ws + oRSUM);

  // bf16 conversions
  conv_f32_bf16<<<(BB*TT*DD)/256, 256, 0, stream>>>(x, xbf, BB*TT*DD);
  conv_f32_bf16<<<(VV*DD)/256,    256, 0, stream>>>(vg, vgbf, VV*DD);
  conv_f32_bf16<<<(BB*TS*DD)/256, 256, 0, stream>>>(eq, eqbf, BB*TS*DD);
  conv_eqT_bf16<<<(BB*TS*DD)/256, 256, 0, stream>>>(eq, eqtbf);
  conv_f32_bf16<<<(DD*DD)/256,    256, 0, stream>>>(Wq, wqbf, DD*DD);
  conv_f32_bf16<<<(DD*DD)/256,    256, 0, stream>>>(Wk, wkbf, DD*DD);
  zero_kernel<<<(BB*TT*VV/4)/256, 256, 0, stream>>>(ptrP, BB*TT*VV/4);

  // q = x @ Wq^T + bq   [1024,512]
  gemm_bf16_tn<<<dim3(128, 1), 128, 0, stream>>>(xbf, wqbf, bq, qf,
      BB*TT, DD, DD, 0, 0, 0);
  // k = eq @ Wk^T + bk  [4096,512]
  gemm_bf16_tn<<<dim3(512, 1), 128, 0, stream>>>(eqbf, wkbf, bk, kf,
      BB*TS, DD, DD, 0, 0, 0);
  conv_f32_bf16<<<(BB*TT*DD)/256, 256, 0, stream>>>(qf, qbf, BB*TT*DD);
  conv_f32_bf16<<<(BB*TS*DD)/256, 256, 0, stream>>>(kf, kbf, BB*TS*DD);

  // scores[b] = q[b] @ k[b]^T  [8][128,512]
  gemm_bf16_tn<<<dim3(16, 8), 128, 0, stream>>>(qbf, kbf, nullptr, sc,
      TT, TS, DD, (long long)TT*DD, (long long)TS*DD, (long long)TT*TS);
  attn_softmax_kernel<<<BB*TT, 256, 0, stream>>>(sc, abf, qmask);

  // text_vec[b] = attn[b] @ eqT[b]^T  [8][128,512]
  gemm_bf16_tn<<<dim3(16, 8), 128, 0, stream>>>(abf, eqtbf, nullptr, tv,
      TT, DD, TS, (long long)TT*TS, (long long)TS*DD, (long long)TT*DD);
  switch_kernel<<<BB*TT, 256, 0, stream>>>(x, tv, etgt, Wg, bg, sw);

  // vocab logits -> d_out  [1024, 32000]
  gemm_bf16_tn<<<dim3(8000, 1), 128, 0, stream>>>(xbf, vgbf, nullptr, out,
      BB*TT, VV, DD, 0, 0, 0);
  vocab_stats_kernel<<<BB*TT, 256, 0, stream>>>(out, rmax, rsum);
  scatter_kernel<<<(BB*TT*TS)/256, 256, 0, stream>>>(sc, query, ptrP);
  combine_kernel<<<(BB*TT*VV/4)/256, 256, 0, stream>>>(out, ptrP, rmax, rsum, sw);
}
